// ScanRNN_1872605741586
// MI455X (gfx1250) — matrix operand order repaired and checked by reading, not compiled, not run on hardware
//
#include <hip/hip_runtime.h>
#include <stdint.h>

// GRU scan for MI455X (gfx1250, wave32, WMMA f32_16x16x32_f16).
// T=2048, B=512, H=128. Each workgroup owns a 16-row batch tile and runs the
// whole time scan privately (recurrence is independent per batch element).
// 8 waves per block; wave w computes gate columns [16w,16w+16) of r/z/n.
// v3: time loop unrolled x2 with ping-pong x-fragment sets so x_t loads +
//     f32->f16 conversion are pipelined one step ahead (off the serial
//     h-chain); LDS parity is compile-time constant.

typedef __attribute__((ext_vector_type(16))) _Float16 v16h;
typedef __attribute__((ext_vector_type(8)))  float    v8f;

#define T_STEPS 2048
#define BATCH   512
#define HDIM    128
#define G3      384     // 3*H
#define BT      16      // batch tile rows per workgroup
#define LDS_ROW 136     // padded row stride (halfs): 272B -> conflict-free b128 reads
#define HBUF_N  (BT * LDS_ROW)

// ---- fragment loaders -------------------------------------------------------

// A-matrix 16x32 f16 layout: lanes 0-15 hold row M=lane, K = kb+{0..7} then
// kb+16+{0..7}; lanes 16-31 hold same rows with K offset +8.
__device__ __forceinline__ v16h load_afrag_x(const float* __restrict__ xrow,
                                             int kb, int hl) {
  const float* p0 = xrow + kb + hl * 8;
  const float* p1 = xrow + kb + 16 + hl * 8;
  v16h a;
#pragma unroll
  for (int j = 0; j < 8; ++j) a[j] = (_Float16)p0[j];
#pragma unroll
  for (int j = 0; j < 8; ++j) a[8 + j] = (_Float16)p1[j];
  return a;
}

__device__ __forceinline__ v16h load_afrag_h(const _Float16* __restrict__ hrow,
                                             int kb, int hl) {
  const _Float16* p0 = hrow + kb + hl * 8;
  const _Float16* p1 = hrow + kb + 16 + hl * 8;
  v16h a;
#pragma unroll
  for (int j = 0; j < 8; ++j) a[j] = p0[j];
#pragma unroll
  for (int j = 0; j < 8; ++j) a[8 + j] = p1[j];
  return a;
}

// B-matrix 32x16 f16 layout: lanes 0-15 hold col N=lane, K = kb+{0..7} then
// kb+16+{0..7}; lanes 16-31 hold col N=lane-16 with K offset +8. W is (H=K, 384) row-major.
__device__ __forceinline__ v16h load_bfrag(const float* __restrict__ W,
                                           int n0, int kb, int hl, int ln) {
  v16h b;
  const int col   = n0 + ln;
  const int kbase = kb + hl * 8;
#pragma unroll
  for (int j = 0; j < 16; ++j)
    b[j] = (_Float16)W[(size_t)(kbase + (j & 7) + (j >> 3) * 16) * G3 + col];
  return b;
}

__device__ __forceinline__ v8f wmma(v16h a, v16h b, v8f c) {
  return __builtin_amdgcn_wmma_f32_16x16x32_f16(false, a, false, b,
                                                (short)0, c, false, false);
}

// Native CDNA5 transcendental tanh. TRANS co-exec hazard is not HW-detected
// (ISA 7.4#3) and inline asm hides the op from the compiler's model, so keep
// one v_nop glued after it.
__device__ __forceinline__ float tanh_hw(float x) {
  float y;
  asm("v_tanh_f32 %0, %1\n\tv_nop" : "=v"(y) : "v"(x));
  return y;
}

__device__ __forceinline__ float sigmoid_hw(float x) {
  return __builtin_fmaf(0.5f, tanh_hw(0.5f * x), 0.5f);
}

// ---- kernel -----------------------------------------------------------------

__launch_bounds__(256, 1)
__global__ void gru_scan_kernel(const float*   __restrict__ xs,      // (T,B,H)
                                const uint8_t* __restrict__ resets,  // (T,B) bool
                                const float*   __restrict__ h0,      // (B,H)
                                const float*   __restrict__ Wi,      // (H,3H)
                                const float*   __restrict__ bi,      // (3H,)
                                const float*   __restrict__ Wh,      // (H,3H)
                                const float*   __restrict__ bhn,     // (H,)
                                float*         __restrict__ out)     // (T,B,H)
{
  __shared__ _Float16 hbuf[2 * HBUF_N];   // double-buffered reset-applied h_t

  const int lane = threadIdx.x & 31;
  const int wave = threadIdx.x >> 5;        // 0..7 -> gate column tile
  const int hl   = lane >> 4;               // lane half
  const int ln   = lane & 15;
  const int r0   = blockIdx.x * BT;         // batch row base

  const int nr = wave * 16;                 // r-gate column base
  const int nz = HDIM + wave * 16;          // z-gate column base
  const int nn = 2 * HDIM + wave * 16;      // n-gate column base

  // Persistent weight B-fragments (held in VGPRs across all 2048 steps).
  v16h bir[4], biz[4], bin[4], bhr[4], bhz[4], bhnf[4];
#pragma unroll
  for (int kk = 0; kk < 4; ++kk) {
    const int kb = kk * 32;
    bir[kk]  = load_bfrag(Wi, nr, kb, hl, ln);
    biz[kk]  = load_bfrag(Wi, nz, kb, hl, ln);
    bin[kk]  = load_bfrag(Wi, nn, kb, hl, ln);
    bhr[kk]  = load_bfrag(Wh, nr, kb, hl, ln);
    bhz[kk]  = load_bfrag(Wh, nz, kb, hl, ln);
    bhnf[kk] = load_bfrag(Wh, nn, kb, hl, ln);
  }

  // Per-lane gate biases (constant over C's 8 rows: N = wave*16 + ln).
  const float b_r  = bi[nr + ln];
  const float b_z  = bi[nz + ln];
  const float b_xn = bi[nn + ln];
  const float b_hn = bhn[wave * 16 + ln];

  // Init: reset-applied h0 (using resets[0]) into buffer 0 and hprev regs.
  // hprev is this wave's output columns in C layout (row i + 8*hl, col ln).
  v8f hprev;
  {
    const uint8_t* rp0 = resets + r0;       // t = 0
#pragma unroll
    for (int i = 0; i < 8; ++i) {
      const int row = i + 8 * hl;
      hprev[i] = rp0[row] ? 0.0f
                          : h0[(size_t)(r0 + row) * HDIM + wave * 16 + ln];
    }
    const int base = threadIdx.x * 8;       // 256 threads x 8 = 16x128
#pragma unroll
    for (int j = 0; j < 8; ++j) {
      const int idx = base + j;
      const int row = idx >> 7;
      const int col = idx & 127;
      const float v = rp0[row] ? 0.0f : h0[(size_t)(r0 + row) * HDIM + col];
      hbuf[row * LDS_ROW + col] = (_Float16)v;
    }
  }

  // Prime the x pipeline: fragments for t = 0.
  v16h xa0[4], xa1[4];
  {
    const float* xrow = xs + (size_t)(r0 + ln) * HDIM;
#pragma unroll
    for (int kk = 0; kk < 4; ++kk) xa0[kk] = load_afrag_x(xrow, kk * 32, hl);
  }
  __syncthreads();

  // One GRU step. parity selects the LDS h buffer (compile-time constant);
  // xaU holds this step's x fragments, xaF is filled for step t+1.
  auto step = [&](int t, int parity, v16h (&xaU)[4], v16h (&xaF)[4]) {
    const _Float16* hrow = hbuf + parity * HBUF_N + ln * LDS_ROW;
    _Float16*       hnxt = hbuf + (parity ^ 1) * HBUF_N;

    // h A-fragments (already reset-applied) — the serial chain head.
    v16h ha[4];
#pragma unroll
    for (int kk = 0; kk < 4; ++kk) ha[kk] = load_afrag_h(hrow, kk * 32, hl);

    // Pipeline: issue x loads for t+1 now; they complete during this step's
    // WMMAs/gates. Clamp keeps the final iteration in-bounds (values unused).
    {
      const int tf = (t + 1 < T_STEPS) ? (t + 1) : t;
      const float* xrow = xs + ((size_t)tf * BATCH + r0 + ln) * HDIM;
#pragma unroll
      for (int kk = 0; kk < 4; ++kk) xaF[kk] = load_afrag_x(xrow, kk * 32, hl);
    }
    if (t + 2 < T_STEPS)
      __builtin_prefetch(xs + ((size_t)(t + 2) * BATCH + r0 + ln) * HDIM, 0, 1);

    // Reset mask for step t+1 (applied at write time).
    const size_t   tn = (t + 1 < T_STEPS) ? (size_t)(t + 1) : (size_t)t;
    const uint8_t* rp = resets + tn * BATCH + r0;
    const uint64_t mm = hl ? *(const uint64_t*)(rp + 8)   // rows 8..15
                           : *(const uint64_t*)(rp);      // rows 0..7

    // Accumulators seeded with per-column biases.
    v8f cr, cz, cxn, chn;
#pragma unroll
    for (int i = 0; i < 8; ++i) { cr[i] = b_r; cz[i] = b_z; cxn[i] = b_xn; chn[i] = b_hn; }

    // 24 WMMAs: r,z fuse x@Wi + h@Wh into one accumulator; n keeps x/h split.
#pragma unroll
    for (int kk = 0; kk < 4; ++kk) {
      cr  = wmma(xaU[kk], bir[kk],  cr);
      cr  = wmma(ha[kk],  bhr[kk],  cr);
      cz  = wmma(xaU[kk], biz[kk],  cz);
      cz  = wmma(ha[kk],  bhz[kk],  cz);
      cxn = wmma(xaU[kk], bin[kk],  cxn);
      chn = wmma(ha[kk],  bhnf[kk], chn);
    }

    // Gates (native v_tanh_f32) + state update + stores.
    // C layout: element i -> (row i + 8*hl, col wave*16 + ln).
#pragma unroll
    for (int i = 0; i < 8; ++i) {
      const float r   = sigmoid_hw(cr[i]);
      const float z   = sigmoid_hw(cz[i]);
      const float nv  = tanh_hw(__builtin_fmaf(r, chn[i], cxn[i]));
      const float hn2 = __builtin_fmaf(z, hprev[i] - nv, nv); // (1-z)*nv + z*h
      const int   row = i + 8 * hl;
      out[((size_t)t * BATCH + r0 + row) * HDIM + wave * 16 + ln] = hn2;
      const float hcar = ((mm >> (8 * i)) & 0xFFu) ? 0.0f : hn2; // reset for t+1
      hprev[i] = hcar;
      hnxt[row * LDS_ROW + wave * 16 + ln] = (_Float16)hcar;
    }

    __syncthreads();  // writes to hnxt visible; safe to overwrite hrow's buffer
  };

  for (int t = 0; t < T_STEPS; t += 2) {
    step(t,     0, xa0, xa1);
    step(t + 1, 1, xa1, xa0);
  }
}

extern "C" void kernel_launch(void* const* d_in, const int* in_sizes, int n_in,
                              void* d_out, int out_size, void* d_ws, size_t ws_size,
                              hipStream_t stream) {
  (void)in_sizes; (void)n_in; (void)out_size; (void)d_ws; (void)ws_size;
  const float*   xs     = (const float*)d_in[0];
  const uint8_t* resets = (const uint8_t*)d_in[1];
  const float*   h0     = (const float*)d_in[2];
  const float*   Wi     = (const float*)d_in[3];
  const float*   bi     = (const float*)d_in[4];
  const float*   Wh     = (const float*)d_in[5];
  const float*   bhn    = (const float*)d_in[6];
  float*         out    = (float*)d_out;

  dim3 grid(BATCH / BT);   // 32 workgroups, one 16-row batch tile each
  dim3 block(256);         // 8 wave32 waves: one per gate-column tile
  gru_scan_kernel<<<grid, block, 0, stream>>>(xs, resets, h0, Wi, bi, Wh, bhn, out);
}
